// GATPred_10574209482836
// MI455X (gfx1250) — matrix operand order repaired and checked by reading, not compiled, not run on hardware
//
#include <hip/hip_runtime.h>

// ---------------- CDNA5 WMMA types ----------------
typedef __attribute__((ext_vector_type(16))) __bf16 v16bf;
typedef __attribute__((ext_vector_type(8)))  float  v8f;
typedef __attribute__((ext_vector_type(4)))  unsigned int u32x4;

__device__ inline __bf16 f2bf(float f) {
  union { unsigned short u; __bf16 b; } cv;
  unsigned int x = __float_as_uint(f);
  cv.u = (unsigned short)((x + 0x7FFFu + ((x >> 16) & 1u)) >> 16);  // RNE
  return cv.b;
}

// ---------------- elementwise helpers ----------------
__global__ void k_cast_bf16(const float* __restrict__ in, __bf16* __restrict__ out, long n) {
  long i = (long)blockIdx.x * blockDim.x + threadIdx.x;
  if (i < n) out[i] = f2bf(in[i]);
}
__global__ void k_fill(float* __restrict__ p, float v, long n) {
  long i = (long)blockIdx.x * blockDim.x + threadIdx.x;
  if (i < n) p[i] = v;
}
__global__ void k_relu(float* __restrict__ p, long n) {
  long i = (long)blockIdx.x * blockDim.x + threadIdx.x;
  if (i < n) { float v = p[i]; p[i] = v > 0.f ? v : 0.f; }
}
__global__ void k_finite0(float* __restrict__ m, long n) {
  long i = (long)blockIdx.x * blockDim.x + threadIdx.x;
  if (i < n) { float v = m[i]; if (!(v > -3.0e38f)) m[i] = 0.f; }
}
__global__ void k_bias_add(float* __restrict__ out, const float* __restrict__ b, long n, int HC) {
  long i = (long)blockIdx.x * blockDim.x + threadIdx.x;
  if (i < n * (long)HC) out[i] += b[i % HC];
}

// ---------------- bf16 WMMA GEMM: C[MxN] = A[MxK] * B[KxN] (row major) ----------------
// Block tile 128x256, 8 waves in 2(M) x 4(N) grid, 64x64 per wave (16 WMMA accums).
// Double-buffered LDS (one barrier per K-chunk); B staged transposed with b32 stores.
#define BM 128
#define BN 256
#define BK 32
#define LDP (BK + 8)   // 40 elements = 80B row stride (16B aligned)

__global__ __launch_bounds__(256) void k_gemm_wmma(
    const __bf16* __restrict__ A, const __bf16* __restrict__ B, float* __restrict__ C,
    int M, int N, int K)
{
  __shared__ __bf16 As[2][BM][LDP];   // [stage][row][k]      20.0 KB
  __shared__ __bf16 Bs[2][BN][LDP];   // [stage][col][k] (T)  40.0 KB

  const int tid  = threadIdx.x;
  const int lane = tid & 31;
  const int wid  = tid >> 5;       // 0..7
  const int wm   = wid & 1;        // M sub-tile: wm*64
  const int wn   = wid >> 1;       // N sub-tile: wn*64
  const int l15  = lane & 15;
  const int lhi  = lane >> 4;

  const int tm0 = blockIdx.x * BM;
  const int tn0 = blockIdx.y * BN;

  v8f acc[4][4];
#pragma unroll
  for (int i = 0; i < 4; i++)
#pragma unroll
    for (int j = 0; j < 4; j++) acc[i][j] = {};

  // A staging: thread -> (row, 16-elem k-half)
  const int lrow  = tid >> 1;          // 0..127
  const int lhalf = (tid & 1) * 16;    // 0 or 16
  // B staging: thread -> (k-pair, 16-col segment)
  const int kp    = (tid >> 4) * 2;    // 0,2,...,30
  const int cseg  = (tid & 15) * 16;   // 0..240

  int grow = tm0 + lrow; if (grow >= M) grow = M - 1;

  auto stage_tiles = [&](int s, int kt) {
    // A tile (128x32): two 16B global loads -> two 16B LDS stores
    const __bf16* a = A + (long)grow * K + kt + lhalf;
    *(u32x4*)(&As[s][lrow][lhalf])     = *(const u32x4*)a;
    *(u32x4*)(&As[s][lrow][lhalf + 8]) = *(const u32x4*)(a + 8);
    if (kt + 2 * BK < K) __builtin_prefetch(a + 2 * BK, 0, 1);
    // B tile (32x256): 2 k-rows x 16 cols per thread; pack k,k+1 into dword stores
    const __bf16* b0 = B + (long)(kt + kp) * N + tn0 + cseg;
    u32x4 r0 = *(const u32x4*)b0;        // k=kp,   cols cseg..cseg+7
    u32x4 r1 = *(const u32x4*)(b0 + 8);  // k=kp,   cols cseg+8..cseg+15
    u32x4 r2 = *(const u32x4*)(b0 + N);  // k=kp+1, cols cseg..cseg+7
    u32x4 r3 = *(const u32x4*)(b0 + N + 8);
    if (kt + 2 * BK < K) __builtin_prefetch(b0 + (long)2 * BK * N, 0, 1);
    const unsigned short* p0 = (const unsigned short*)&r0;
    const unsigned short* p1 = (const unsigned short*)&r1;
    const unsigned short* p2 = (const unsigned short*)&r2;
    const unsigned short* p3 = (const unsigned short*)&r3;
#pragma unroll
    for (int j = 0; j < 8; j++)
      *(unsigned int*)(&Bs[s][cseg + j][kp]) =
          (unsigned int)p0[j] | ((unsigned int)p2[j] << 16);
#pragma unroll
    for (int j = 0; j < 8; j++)
      *(unsigned int*)(&Bs[s][cseg + 8 + j][kp]) =
          (unsigned int)p1[j] | ((unsigned int)p3[j] << 16);
  };

  stage_tiles(0, 0);
  __syncthreads();

  for (int kt = 0; kt < K; kt += BK) {
    const int s = (kt / BK) & 1;
    if (kt + BK < K) stage_tiles(s ^ 1, kt + BK);

    // A 16x32 fragment: lanes0-15 K=0..15, lanes16-31 K=16..31;
    // elements 0..15 -> K base+0..15
    v16bf af[4];
#pragma unroll
    for (int mi = 0; mi < 4; mi++) {
      const __bf16* ap = &As[s][wm * 64 + mi * 16 + l15][lhi * 16];
#pragma unroll
      for (int i = 0; i < 8; i++) { af[mi][i] = ap[i]; af[mi][8 + i] = ap[8 + i]; }
    }
    // B 32x16 fragment: lanes0-15 N=lane K=0..15, lanes16-31 N=lane-16 K=16..31
    v16bf bfv[4];
#pragma unroll
    for (int ni = 0; ni < 4; ni++) {
      const __bf16* bp = &Bs[s][wn * 64 + ni * 16 + l15][lhi * 16];
#pragma unroll
      for (int i = 0; i < 16; i++) bfv[ni][i] = bp[i];
    }
#pragma unroll
    for (int mi = 0; mi < 4; mi++)
#pragma unroll
      for (int ni = 0; ni < 4; ni++)
        acc[mi][ni] = __builtin_amdgcn_wmma_f32_16x16x32_bf16(
            false, af[mi], false, bfv[ni], (short)0, acc[mi][ni], false, false);
    __syncthreads();
  }

  // C/D layout: VGPR r: lanes0-15 -> M=r, lanes16-31 -> M=r+8; N = lane&15
#pragma unroll
  for (int mi = 0; mi < 4; mi++) {
#pragma unroll
    for (int ni = 0; ni < 4; ni++) {
      int r0 = tm0 + wm * 64 + mi * 16 + lhi * 8;
      int c  = tn0 + wn * 64 + ni * 16 + l15;
#pragma unroll
      for (int r = 0; r < 8; r++) {
        int row = r0 + r;
        if (row < M) C[(long)row * N + c] = acc[mi][ni][r];
      }
    }
  }
}

// ---------------- attention kernels ----------------
__global__ void k_rowdot(const float* __restrict__ h, const float* __restrict__ avec,
                         float* __restrict__ out, int n, int H, int C) {
  long i = (long)blockIdx.x * blockDim.x + threadIdx.x;
  if (i >= (long)n * H) return;
  int node = (int)(i / H), hh = (int)(i % H);
  const float* hp = h + (long)node * H * C + (long)hh * C;
  const float* ap = avec + (long)hh * C;
  float s = 0.f;
  for (int c = 0; c < C; c++) s += hp[c] * ap[c];
  out[i] = s;
}

__device__ inline void atomicMaxF(float* addr, float val) {
  if (val >= 0.f) atomicMax((int*)addr, __float_as_int(val));
  else            atomicMin((unsigned int*)addr, __float_as_uint(val));
}

__global__ void k_edge_max(const int* __restrict__ ei, int E,
                           const float* __restrict__ a_s, const float* __restrict__ a_d,
                           float* __restrict__ ebuf, float* __restrict__ m, int H) {
  long i = (long)blockIdx.x * blockDim.x + threadIdx.x;
  if (i >= (long)E * H) return;
  int e = (int)(i / H), h = (int)(i % H);
  int src = ei[e], dst = ei[E + e];
  float v = a_s[(long)src * H + h] + a_d[(long)dst * H + h];
  v = v > 0.f ? v : 0.2f * v;      // leaky_relu(0.2)
  ebuf[i] = v;
  atomicMaxF(&m[(long)dst * H + h], v);
}

__global__ void k_edge_exp(const int* __restrict__ ei, int E,
                           float* __restrict__ ebuf, const float* __restrict__ m,
                           float* __restrict__ den, int H) {
  long i = (long)blockIdx.x * blockDim.x + threadIdx.x;
  if (i >= (long)E * H) return;
  int e = (int)(i / H), h = (int)(i % H);
  int dst = ei[E + e];
  float ex = __expf(ebuf[i] - m[(long)dst * H + h]);
  ebuf[i] = ex;
  atomicAdd(&den[(long)dst * H + h], ex);
}

__global__ __launch_bounds__(256) void k_edge_scatter(
    const int* __restrict__ ei, int E, const float* __restrict__ ebuf,
    const float* __restrict__ den, const float* __restrict__ hs,
    float* __restrict__ out, int H, int C) {
  int e = blockIdx.x;
  int src = ei[e], dst = ei[E + e];
  int HC = H * C;
  for (int idx = threadIdx.x; idx < HC; idx += blockDim.x) {
    int h = idx / C;
    float alpha = ebuf[(long)e * H + h] / (den[(long)dst * H + h] + 1e-16f);
    atomicAdd(&out[(long)dst * HC + idx], alpha * hs[(long)src * HC + idx]);
  }
}

__global__ __launch_bounds__(256) void k_final_dot(
    const int* __restrict__ eli, int E, const float* __restrict__ hu,
    const float* __restrict__ hp, float* __restrict__ out) {
  int pidx = blockIdx.x;
  int u = eli[pidx], v = eli[E + pidx];
  const float* a = hu + (long)u * 768;
  const float* b = hp + (long)v * 768;
  float s = 0.f;
  for (int i = threadIdx.x; i < 768; i += 256) s += a[i] * b[i];
#pragma unroll
  for (int off = 16; off > 0; off >>= 1) s += __shfl_down(s, off, 32);
  __shared__ float red[8];
  if ((threadIdx.x & 31) == 0) red[threadIdx.x >> 5] = s;
  __syncthreads();
  if (threadIdx.x == 0) {
    float t = 0.f;
#pragma unroll
    for (int w = 0; w < 8; w++) t += red[w];
    out[pidx] = t;
  }
}

// ---------------- host orchestration ----------------
extern "C" void kernel_launch(void* const* d_in, const int* in_sizes, int n_in,
                              void* d_out, int out_size, void* d_ws, size_t ws_size,
                              hipStream_t stream) {
  (void)n_in; (void)ws_size; (void)out_size;
  const int NS[4] = {10000, 20000, 5000, 2000};  // user, paper, method, task
  const float* x[4] = {(const float*)d_in[0], (const float*)d_in[1],
                       (const float*)d_in[2], (const float*)d_in[3]};

  char* wp = (char*)d_ws;
  auto take = [&](size_t bytes) { char* r = wp; wp += (bytes + 255) & ~(size_t)255; return r; };
  __bf16* xb[4];  for (int t = 0; t < 4; t++) xb[t]  = (__bf16*)take((size_t)NS[t] * 768 * 2);
  float*  h1[4];  for (int t = 0; t < 4; t++) h1[t]  = (float*) take((size_t)NS[t] * 1024 * 4);
  __bf16* h1b[4]; for (int t = 0; t < 4; t++) h1b[t] = (__bf16*)take((size_t)NS[t] * 1024 * 2);
  float* h2u = (float*)take((size_t)NS[0] * 768 * 4);
  float* h2p = (float*)take((size_t)NS[1] * 768 * 4);
  float* hsb = (float*)take((size_t)20000 * 1024 * 4);
  float* hdb = (float*)take((size_t)20000 * 1024 * 4);
  __bf16* Wb = (__bf16*)take((size_t)1024 * 1024 * 2);
  float* asb = (float*)take((size_t)20000 * 8 * 4);
  float* adb = (float*)take((size_t)20000 * 8 * 4);
  float* mb  = (float*)take((size_t)20000 * 8 * 4);
  float* db  = (float*)take((size_t)20000 * 8 * 4);
  float* eb  = (float*)take((size_t)80000 * 8 * 4);

  auto cdiv = [](long a, long b) { return (int)((a + b - 1) / b); };
  const float NEG_INF = -__builtin_huge_valf();

  // One GAT conv for one edge type. Params flattened per-jax-tree:
  // d_in[12 + 4*pj + {0:W, 1:a_dst, 2:a_src, 3:b}]
  auto launch_conv = [&](int ei_idx, int pj, const __bf16* xs_b, const __bf16* xd_b,
                         int n_src, int n_dst, float* outacc, int H, int C, int K) {
    const float* W  = (const float*)d_in[12 + 4 * pj + 0];
    const float* ad = (const float*)d_in[12 + 4 * pj + 1];
    const float* as = (const float*)d_in[12 + 4 * pj + 2];
    const float* bv = (const float*)d_in[12 + 4 * pj + 3];
    const int* ei = (const int*)d_in[ei_idx];
    int E = in_sizes[ei_idx] / 2;
    int HC = H * C;

    k_cast_bf16<<<cdiv((long)K * HC, 256), 256, 0, stream>>>(W, Wb, (long)K * HC);
    { dim3 g(cdiv(n_src, BM), HC / BN);
      k_gemm_wmma<<<g, 256, 0, stream>>>(xs_b, Wb, hsb, n_src, HC, K); }
    const float* hd = hsb;
    if (xd_b != xs_b) {
      dim3 g(cdiv(n_dst, BM), HC / BN);
      k_gemm_wmma<<<g, 256, 0, stream>>>(xd_b, Wb, hdb, n_dst, HC, K);
      hd = hdb;
    }
    k_rowdot<<<cdiv((long)n_src * H, 256), 256, 0, stream>>>(hsb, as, asb, n_src, H, C);
    k_rowdot<<<cdiv((long)n_dst * H, 256), 256, 0, stream>>>(hd,  ad, adb, n_dst, H, C);
    k_fill<<<cdiv((long)n_dst * H, 256), 256, 0, stream>>>(mb, NEG_INF, (long)n_dst * H);
    k_fill<<<cdiv((long)n_dst * H, 256), 256, 0, stream>>>(db, 0.f,     (long)n_dst * H);
    k_edge_max<<<cdiv((long)E * H, 256), 256, 0, stream>>>(ei, E, asb, adb, eb, mb, H);
    k_finite0<<<cdiv((long)n_dst * H, 256), 256, 0, stream>>>(mb, (long)n_dst * H);
    k_edge_exp<<<cdiv((long)E * H, 256), 256, 0, stream>>>(ei, E, eb, mb, db, H);
    k_edge_scatter<<<E, 256, 0, stream>>>(ei, E, eb, db, hsb, outacc, H, C);
    k_bias_add<<<cdiv((long)n_dst * HC, 256), 256, 0, stream>>>(outacc, bv, (long)n_dst, HC);
  };

  // ---- layer 1 (H=8, C=128, K=768) ----
  for (int t = 0; t < 4; t++)
    k_cast_bf16<<<cdiv((long)NS[t] * 768, 256), 256, 0, stream>>>(x[t], xb[t], (long)NS[t] * 768);
  for (int t = 0; t < 4; t++)
    k_fill<<<cdiv((long)NS[t] * 1024, 256), 256, 0, stream>>>(h1[t], 0.f, (long)NS[t] * 1024);

  // {ei input idx, l1 param idx, src type, dst type}; node types: 0=user,1=paper,2=method,3=task
  struct EtL1 { int ei, pj, st, dt; };
  const EtL1 ets1[7] = {
    {4, 1, 1, 1},   // cites         paper->paper
    {5, 0, 1, 2},   // applies       paper->method
    {6, 4, 2, 1},   // rev_applies   method->paper
    {7, 3, 1, 3},   // performs      paper->task
    {8, 6, 3, 1},   // rev_performs  task->paper
    {9, 2, 0, 1},   // likes         user->paper
    {10, 5, 1, 0},  // rev_likes     paper->user
  };
  for (int i = 0; i < 7; i++) {
    const EtL1& e = ets1[i];
    launch_conv(e.ei, e.pj, xb[e.st], xb[e.dt], NS[e.st], NS[e.dt], h1[e.dt], 8, 128, 768);
  }
  for (int t = 0; t < 4; t++) {
    long n = (long)NS[t] * 1024;
    k_relu<<<cdiv(n, 256), 256, 0, stream>>>(h1[t], n);
    k_cast_bf16<<<cdiv(n, 256), 256, 0, stream>>>(h1[t], h1b[t], n);
  }

  // ---- layer 2 (H=1, C=768, K=1024); only dst user/paper feed the output ----
  k_fill<<<cdiv((long)NS[0] * 768, 256), 256, 0, stream>>>(h2u, 0.f, (long)NS[0] * 768);
  k_fill<<<cdiv((long)NS[1] * 768, 256), 256, 0, stream>>>(h2p, 0.f, (long)NS[1] * 768);

  struct EtL2 { int ei, pj, st, dt; float* acc; };
  const EtL2 ets2[5] = {
    {4,  8,  1, 1, h2p},   // cites
    {6,  11, 2, 1, h2p},   // rev_applies
    {8,  13, 3, 1, h2p},   // rev_performs
    {9,  9,  0, 1, h2p},   // likes
    {10, 12, 1, 0, h2u},   // rev_likes
  };
  for (int i = 0; i < 5; i++) {
    const EtL2& e = ets2[i];
    launch_conv(e.ei, e.pj, h1b[e.st], h1b[e.dt], NS[e.st], NS[e.dt], e.acc, 1, 768, 1024);
  }

  // ---- link prediction dot products ----
  const int* eli = (const int*)d_in[11];
  int EP = in_sizes[11] / 2;
  k_final_dot<<<EP, 256, 0, stream>>>(eli, EP, h2u, h2p, (float*)d_out);
}
